// GNNEncoder_70918499992071
// MI455X (gfx1250) — matrix operand order repaired and checked by reading, not compiled, not run on hardware
//
#include <hip/hip_runtime.h>
#include <hip/hip_bf16.h>

#define NEG_SLOPE 0.2f
#define LN_EPS 1e-5f

typedef __bf16 bfv16 __attribute__((ext_vector_type(16)));
typedef float  fv8   __attribute__((ext_vector_type(8)));

__device__ __forceinline__ unsigned short f2bf(float f) {
    unsigned int u = __float_as_uint(f);
    unsigned int r = u + 0x7FFFu + ((u >> 16) & 1u);   // round-to-nearest-even
    return (unsigned short)(r >> 16);
}
// order-preserving float<->int encoding for atomicMax-based segment max
__device__ __forceinline__ int encf(float f) {
    int i = __float_as_int(f);
    return (i >= 0) ? i : (i ^ 0x7FFFFFFF);
}
__device__ __forceinline__ float decf(int i) {
    return __int_as_float((i >= 0) ? i : (i ^ 0x7FFFFFFF));
}
// CDNA5 ISA 7.12.2: 16-bit A 16x32 fragment K-index for (lane-half h, elem i)
__device__ __forceinline__ int a_kmap(int h, int i) {
    int v = i >> 1, odd = i & 1;
    return ((v < 4) ? (h * 8 + 2 * v) : (16 + h * 8 + 2 * (v - 4))) + odd;
}
// 16-bit B 32x16 fragment K-index
__device__ __forceinline__ int b_kmap(int h, int i) {
    int v = i >> 1, odd = i & 1;
    return ((v < 4) ? (h * 8 + 2 * v) : (16 + h * 8 + 2 * (v - 4))) + odd;
}

// ---------------- fills ----------------
__global__ void fill_f32(float* __restrict__ p, float v, int n) {
    int i = blockIdx.x * blockDim.x + threadIdx.x;
    if (i < n) p[i] = v;
}
__global__ void fill_i32(int* __restrict__ p, int v, int n) {
    int i = blockIdx.x * blockDim.x + threadIdx.x;
    if (i < n) p[i] = v;
}

// ---------------- fragment packing ----------------
// layer-1 A: x[N,5] f32 -> Apack[tile][lane][16] bf16 (K zero-padded 5->32, 1 k-tile)
__global__ void pack_A_from_x(const float* __restrict__ x, unsigned short* __restrict__ Ap,
                              int n, int nTiles) {
    int t = blockIdx.x * blockDim.x + threadIdx.x;
    if (t >= nTiles * 32) return;
    int tile = t >> 5, lane = t & 31;
    int h = lane >> 4;
    int node = tile * 16 + (lane & 15);
    if (node >= n) node = n - 1;
    unsigned short* out = Ap + (size_t)t * 16;
#pragma unroll
    for (int i = 0; i < 16; ++i) {
        int kk = a_kmap(h, i);
        out[i] = (kk < 5) ? f2bf(x[(size_t)node * 5 + kk]) : (unsigned short)0;
    }
}
// generic A repack: src bf16 [n,K] row-major -> Apack[tile][kt][lane][16]
__global__ void pack_A_frag(const unsigned short* __restrict__ src, unsigned short* __restrict__ Ap,
                            int n, int K, int nTiles, int nKt) {
    int t = blockIdx.x * blockDim.x + threadIdx.x;
    if (t >= nTiles * nKt * 32) return;
    int tile = t / (nKt * 32);
    int rem  = t % (nKt * 32);
    int kt   = rem >> 5;
    int lane = rem & 31;
    int h = lane >> 4;
    int node = tile * 16 + (lane & 15);
    if (node >= n) node = n - 1;
    const unsigned short* row = src + (size_t)node * K + kt * 32;
    unsigned short* out = Ap + (size_t)t * 16;
#pragma unroll
    for (int i = 0; i < 16; ++i) out[i] = row[a_kmap(h, i)];
}
// B pack: [Wl|Wr] (f32, Kr x C each) -> Bpack[kt][nt][lane][16] bf16, rows >= Kr zero
__global__ void pack_B_frag(const float* __restrict__ Wl, const float* __restrict__ Wr,
                            unsigned short* __restrict__ Bp, int Kr, int C, int nct, int nKt) {
    int t = blockIdx.x * blockDim.x + threadIdx.x;
    if (t >= nKt * nct * 32) return;
    int kt   = t / (nct * 32);
    int rem  = t % (nct * 32);
    int nt   = rem >> 5;
    int lane = rem & 31;
    int h = lane >> 4;
    int col = nt * 16 + (lane & 15);
    unsigned short* out = Bp + (size_t)t * 16;
#pragma unroll
    for (int i = 0; i < 16; ++i) {
        int kk = kt * 32 + b_kmap(h, i);
        float v = 0.f;
        if (kk < Kr) v = (col < C) ? Wl[(size_t)kk * C + col] : Wr[(size_t)kk * C + (col - C)];
        out[i] = f2bf(v);
    }
}

// ---------------- WMMA GEMM on pre-packed fragments ----------------
// O[Npad, 2*COUT] = A x W -> outL/outR (each padded to nTiles*16 rows, so no tail guards).
// Column tiles 0..NCT/2-1 -> outL, NCT/2..NCT-1 -> outR (compile-time split).
template <int NCT, int NKT, int COUT>
__global__ void gemm_bf16_wmma(const unsigned short* __restrict__ Ap,
                               const unsigned short* __restrict__ Bp,
                               float* __restrict__ outL, float* __restrict__ outR,
                               int nTiles) {
    static_assert((NCT / 2) * 16 == COUT, "tile split must match COUT");
    const int lane = threadIdx.x & 31;
    const int wave = threadIdx.x >> 5;
    const int tile = blockIdx.x * (blockDim.x >> 5) + wave;
    if (tile >= nTiles) return;                 // uniform per wave: EXEC stays all-1s
    const int r = lane & 15;                    // D column within tile
    const int h = lane >> 4;

    fv8 acc[NCT];
    const fv8 zero = {0.f, 0.f, 0.f, 0.f, 0.f, 0.f, 0.f, 0.f};
#pragma unroll
    for (int t = 0; t < NCT; ++t) acc[t] = zero;

#pragma unroll
    for (int kt = 0; kt < NKT; ++kt) {
        bfv16 a = *reinterpret_cast<const bfv16*>(
            Ap + ((size_t)(tile * NKT + kt) * 32 + lane) * 16);
#pragma unroll
        for (int t = 0; t < NCT; ++t) {
            bfv16 b = *reinterpret_cast<const bfv16*>(
                Bp + ((size_t)(kt * NCT + t) * 32 + lane) * 16);
            acc[t] = __builtin_amdgcn_wmma_f32_16x16x32_bf16(
                false, a, false, b, (short)0, acc[t], false, false);
        }
    }
    // epilogue: outputs are padded to nTiles*16 rows -> unconditional stores
    const size_t rowOff = (size_t)(tile * 16 + h * 8) * COUT;
#pragma unroll
    for (int t = 0; t < NCT; ++t) {
        float* outp = ((t < NCT / 2) ? (outL + (t * 16 + r))
                                     : (outR + ((t - NCT / 2) * 16 + r))) + rowOff;
#pragma unroll
        for (int v = 0; v < 8; ++v) outp[v * COUT] = acc[t][v];
    }
}

// ---------------- edge passes ----------------
// pass 1: logits[e,h] = att[h,:] . leakyrelu(xl[src]+xr[dst]); segment-max into lmaxEnc[dst,h]
template <int H, int C>
__global__ void edge_logits(const long long* __restrict__ esrc, const long long* __restrict__ edst,
                            const float* __restrict__ xl, const float* __restrict__ xr,
                            const float* __restrict__ att,
                            float* __restrict__ logits, int* __restrict__ lmaxEnc,
                            int E, int Etot) {
    int e = blockIdx.x * blockDim.x + threadIdx.x;
    if (e >= Etot) return;
    int s, d;
    if (e < E) { s = (int)esrc[e]; d = (int)edst[e]; }
    else       { s = d = e - E; }                       // self-loop
    const float* pl = xl + (size_t)s * H * C;
    const float* pr = xr + (size_t)d * H * C;
#pragma unroll
    for (int hh = 0; hh < H; ++hh) {
        float acc = 0.f;
#pragma unroll
        for (int c = 0; c < C; ++c) {
            float v = pl[hh * C + c] + pr[hh * C + c];
            v = (v > 0.f) ? v : NEG_SLOPE * v;
            acc += att[hh * C + c] * v;
        }
        logits[(size_t)e * H + hh] = acc;
        atomicMax(&lmaxEnc[d * H + hh], encf(acc));
    }
}
// pass 2: p = exp(logit - lmax[dst]); denom[dst,h] += p; store p in place
template <int H>
__global__ void edge_softmax_p(const long long* __restrict__ edst,
                               float* __restrict__ logits, const int* __restrict__ lmaxEnc,
                               float* __restrict__ denom, int E, int Etot) {
    int e = blockIdx.x * blockDim.x + threadIdx.x;
    if (e >= Etot) return;
    int d = (e < E) ? (int)edst[e] : (e - E);
#pragma unroll
    for (int hh = 0; hh < H; ++hh) {
        float l = logits[(size_t)e * H + hh];
        float p = __expf(l - decf(lmaxEnc[d * H + hh]));
        logits[(size_t)e * H + hh] = p;
        atomicAdd(&denom[d * H + hh], p);
    }
}
// pass 3: agg[dst] += xl[src] * (p / denom[dst])
template <int H, int C>
__global__ void edge_agg(const long long* __restrict__ esrc, const long long* __restrict__ edst,
                         const float* __restrict__ xl, const float* __restrict__ logits,
                         const float* __restrict__ denom, float* __restrict__ agg,
                         int E, int Etot) {
    int e = blockIdx.x * blockDim.x + threadIdx.x;
    if (e >= Etot) return;
    int s, d;
    if (e < E) { s = (int)esrc[e]; d = (int)edst[e]; }
    else       { s = d = e - E; }
    const float* pl = xl + (size_t)s * H * C;
    float* pa = agg + (size_t)d * H * C;
#pragma unroll
    for (int hh = 0; hh < H; ++hh) {
        float alpha = logits[(size_t)e * H + hh] / denom[d * H + hh];
#pragma unroll
        for (int c = 0; c < C; ++c)
            atomicAdd(&pa[hh * C + c], pl[hh * C + c] * alpha);
    }
}

// ---------------- layernorms (wave-per-node, wave32 shuffle reductions) ----------------
__global__ void ln1_elu_tobf(const float* __restrict__ agg, const float* __restrict__ b,
                             const float* __restrict__ g, const float* __restrict__ beta,
                             unsigned short* __restrict__ hbf, int n) {
    int node = blockIdx.x * (blockDim.x >> 5) + (threadIdx.x >> 5);
    int lane = threadIdx.x & 31;
    if (node >= n) return;
    const float* row = agg + (size_t)node * 64;
    float v0 = row[lane] + b[lane];
    float v1 = row[lane + 32] + b[lane + 32];
    float s = v0 + v1;
#pragma unroll
    for (int m = 16; m >= 1; m >>= 1) s += __shfl_xor(s, m, 32);
    float mu = s * (1.0f / 64.0f);
    float d0 = v0 - mu, d1 = v1 - mu;
    float q = d0 * d0 + d1 * d1;
#pragma unroll
    for (int m = 16; m >= 1; m >>= 1) q += __shfl_xor(q, m, 32);
    float rs = rsqrtf(q * (1.0f / 64.0f) + LN_EPS);
    float y0 = d0 * rs * g[lane] + beta[lane];
    float y1 = d1 * rs * g[lane + 32] + beta[lane + 32];
    y0 = (y0 > 0.f) ? y0 : (__expf(y0) - 1.f);   // ELU
    y1 = (y1 > 0.f) ? y1 : (__expf(y1) - 1.f);
    hbf[(size_t)node * 64 + lane]      = f2bf(y0);
    hbf[(size_t)node * 64 + lane + 32] = f2bf(y1);
}

__global__ void ln2_out(const float* __restrict__ agg, const float* __restrict__ b,
                        const float* __restrict__ g, const float* __restrict__ beta,
                        float* __restrict__ out, int n) {
    int node = blockIdx.x * (blockDim.x >> 5) + (threadIdx.x >> 5);
    int lane = threadIdx.x & 31;
    if (node >= n) return;
    float v = agg[(size_t)node * 32 + lane] + b[lane];
    float s = v;
#pragma unroll
    for (int m = 16; m >= 1; m >>= 1) s += __shfl_xor(s, m, 32);
    float mu = s * (1.0f / 32.0f);
    float d = v - mu;
    float q = d * d;
#pragma unroll
    for (int m = 16; m >= 1; m >>= 1) q += __shfl_xor(q, m, 32);
    float rs = rsqrtf(q * (1.0f / 32.0f) + LN_EPS);
    out[(size_t)node * 32 + lane] = d * rs * g[lane] + beta[lane];
}

// ---------------- host launch ----------------
extern "C" void kernel_launch(void* const* d_in, const int* in_sizes, int n_in,
                              void* d_out, int out_size, void* d_ws, size_t ws_size,
                              hipStream_t stream) {
    (void)n_in; (void)out_size; (void)ws_size;
    const float*     x    = (const float*)d_in[0];
    const long long* eidx = (const long long*)d_in[1];   // int64 [2,E]
    const float* Wl1  = (const float*)d_in[2];
    const float* Wr1  = (const float*)d_in[3];
    const float* att1 = (const float*)d_in[4];
    const float* b1   = (const float*)d_in[5];
    const float* g1   = (const float*)d_in[6];
    const float* be1  = (const float*)d_in[7];
    const float* Wl2  = (const float*)d_in[8];
    const float* Wr2  = (const float*)d_in[9];
    const float* att2 = (const float*)d_in[10];
    const float* b2   = (const float*)d_in[11];
    const float* g2   = (const float*)d_in[12];
    const float* be2  = (const float*)d_in[13];

    const int N = in_sizes[0] / 5;
    const int E = in_sizes[1] / 2;
    const int Etot = E + N;
    const long long* esrc = eidx;
    const long long* edst = eidx + E;
    const int nTiles = (N + 15) / 16;
    const int Npad = nTiles * 16;   // GEMM outputs padded so the epilogue needs no guards

    char* base = (char*)d_ws;
    size_t off = 0;
    auto bump = [&](size_t bytes) -> char* {
        char* p = base + off;
        off += (bytes + 255) & ~(size_t)255;
        return p;
    };
    // persistent region
    unsigned short* hbf = (unsigned short*)bump((size_t)N * 64 * 2);
    unsigned short* Ap1 = (unsigned short*)bump((size_t)nTiles * 1 * 32 * 16 * 2);
    unsigned short* Ap2 = (unsigned short*)bump((size_t)nTiles * 2 * 32 * 16 * 2);
    unsigned short* Bp1 = (unsigned short*)bump((size_t)1 * 8 * 32 * 16 * 2);
    unsigned short* Bp2 = (unsigned short*)bump((size_t)2 * 4 * 32 * 16 * 2);
    size_t offShared = off;
    // layer-1 overlay
    float* xl1  = (float*)bump((size_t)Npad * 64 * 4);
    float* xr1  = (float*)bump((size_t)Npad * 64 * 4);
    float* agg1 = (float*)bump((size_t)N * 64 * 4);
    float* lg1  = (float*)bump((size_t)Etot * 4 * 4);
    int*   lm1  = (int*)  bump((size_t)N * 4 * 4);
    float* dn1  = (float*)bump((size_t)N * 4 * 4);
    // layer-2 overlay (layer-1 edge buffers dead after ln1)
    off = offShared;
    float* xl2  = (float*)bump((size_t)Npad * 32 * 4);
    float* xr2  = (float*)bump((size_t)Npad * 32 * 4);
    float* agg2 = (float*)bump((size_t)N * 32 * 4);
    float* lg2  = (float*)bump((size_t)Etot * 4);
    int*   lm2  = (int*)  bump((size_t)N * 4);
    float* dn2  = (float*)bump((size_t)N * 4);

    const int TB = 256;
    auto cdiv = [](int a, int b) { return (a + b - 1) / b; };

    // operand packing (fragment layouts)
    pack_A_from_x<<<cdiv(nTiles * 32, TB), TB, 0, stream>>>(x, Ap1, N, nTiles);
    pack_B_frag<<<1, 256, 0, stream>>>(Wl1, Wr1, Bp1, 5, 64, 8, 1);
    pack_B_frag<<<1, 256, 0, stream>>>(Wl2, Wr2, Bp2, 64, 32, 4, 2);

    // layer 1: [N,32] x [32,128] -> xl1,xr1   (WMMA)
    gemm_bf16_wmma<8, 1, 64><<<cdiv(nTiles, 4), 128, 0, stream>>>(Ap1, Bp1, xl1, xr1, nTiles);

    fill_i32<<<cdiv(N * 4, TB), TB, 0, stream>>>(lm1, (int)0x80000000, N * 4);
    fill_f32<<<cdiv(N * 4, TB), TB, 0, stream>>>(dn1, 0.f, N * 4);
    fill_f32<<<cdiv(N * 64, TB), TB, 0, stream>>>(agg1, 0.f, N * 64);
    edge_logits<4, 16><<<cdiv(Etot, TB), TB, 0, stream>>>(esrc, edst, xl1, xr1, att1, lg1, lm1, E, Etot);
    edge_softmax_p<4><<<cdiv(Etot, TB), TB, 0, stream>>>(edst, lg1, lm1, dn1, E, Etot);
    edge_agg<4, 16><<<cdiv(Etot, TB), TB, 0, stream>>>(esrc, edst, xl1, lg1, dn1, agg1, E, Etot);
    ln1_elu_tobf<<<cdiv(N, 8), 256, 0, stream>>>(agg1, b1, g1, be1, hbf, N);
    pack_A_frag<<<cdiv(nTiles * 2 * 32, TB), TB, 0, stream>>>(hbf, Ap2, N, 64, nTiles, 2);

    // layer 2 scratch init (safe: layer-1 edge buffers dead after ln1)
    fill_i32<<<cdiv(N, TB), TB, 0, stream>>>(lm2, (int)0x80000000, N);
    fill_f32<<<cdiv(N, TB), TB, 0, stream>>>(dn2, 0.f, N);
    fill_f32<<<cdiv(N * 32, TB), TB, 0, stream>>>(agg2, 0.f, N * 32);
    // layer 2: [N,64] x [64,64] -> xl2,xr2   (WMMA)
    gemm_bf16_wmma<4, 2, 32><<<cdiv(nTiles, 4), 128, 0, stream>>>(Ap2, Bp2, xl2, xr2, nTiles);

    edge_logits<1, 32><<<cdiv(Etot, TB), TB, 0, stream>>>(esrc, edst, xl2, xr2, att2, lg2, lm2, E, Etot);
    edge_softmax_p<1><<<cdiv(Etot, TB), TB, 0, stream>>>(edst, lg2, lm2, dn2, E, Etot);
    edge_agg<1, 32><<<cdiv(Etot, TB), TB, 0, stream>>>(esrc, edst, xl2, lg2, dn2, agg2, E, Etot);
    ln2_out<<<cdiv(N, 8), 256, 0, stream>>>(agg2, b2, g2, be2, (float*)d_out, N);
}
